// IQARegression_2628519985592
// MI455X (gfx1250) — matrix operand order repaired and checked by reading, not compiled, not run on hardware
//
#include <hip/hip_runtime.h>

// ---------------------------------------------------------------------------
// MI455X (gfx1250, wave32) implementation.
// Heavy GEMMs (1x1 conv, 3x3 conv, q/k/v/out projections) run on
// V_WMMA_F32_16X16X32_BF16 with double-buffered LDS staging via
// GLOBAL_LOAD_ASYNC_TO_LDS_B128 (ASYNCcnt) when the toolchain exposes it.
// Precision-sensitive reductions (layernorm, softmax, pooling, MoE) stay
// fp32 VALU.
// ---------------------------------------------------------------------------

typedef __attribute__((ext_vector_type(4)))  unsigned int u32x4;
typedef __attribute__((ext_vector_type(4)))  float        f32x4;
typedef __attribute__((ext_vector_type(8)))  float        v8f;
typedef __attribute__((ext_vector_type(16))) __bf16       v16bf;

union Frag  { u32x4 u[2]; v16bf v; };          // 32B: one WMMA A/B operand per lane
union Pack8 { unsigned short s[8]; u32x4 u; }; // 8 bf16 -> one b128 store

constexpr int KP = 48;   // LDS row pitch in halves (96B): aligned b128 frag reads

#if defined(__AMDGCN__) && \
    __has_builtin(__builtin_amdgcn_global_load_async_to_lds_b128) && \
    __has_builtin(__builtin_amdgcn_s_wait_asynccnt)
#define HAS_ASYNC_LDS 1
#else
#define HAS_ASYNC_LDS 0
#endif

#if HAS_ASYNC_LDS
// Builtin signature (from the compiler diagnostic): pointee is a 16B int
// vector; src is global (AS1), dst is LDS (AS3); trailing imm offset + cpol.
typedef int v4i_vec __attribute__((vector_size(16)));
typedef __attribute__((address_space(1))) v4i_vec* gas_v4i_ptr;
typedef __attribute__((address_space(3))) v4i_vec* las_v4i_ptr;

// one 16B async transfer: global -> LDS, bypassing VGPRs (tracked by ASYNCcnt)
__device__ __forceinline__ void async_cp16(const unsigned short* g, unsigned short* l) {
    __builtin_amdgcn_global_load_async_to_lds_b128(
        (gas_v4i_ptr)(g), (las_v4i_ptr)(l), 0, 0);
}
__device__ __forceinline__ void async_wait0() { __builtin_amdgcn_s_wait_asynccnt(0); }
#endif

__device__ __forceinline__ unsigned short f2bf(float f) {
    unsigned int u = __float_as_uint(f);
    unsigned int r = u + 0x7fffu + ((u >> 16) & 1u);   // round-to-nearest-even
    return (unsigned short)(r >> 16);
}

__device__ __forceinline__ float wred_sum(float v) {
    for (int o = 16; o > 0; o >>= 1) v += __shfl_xor(v, o, 32);
    return v;
}
__device__ __forceinline__ float wred_max(float v) {
    for (int o = 16; o > 0; o >>= 1) v = fmaxf(v, __shfl_xor(v, o, 32));
    return v;
}

// fragment loads per CDNA5 ISA layout + 4x2 WMMA tile accumulate
__device__ __forceinline__ void frag_mma(
    const unsigned short* As, const unsigned short* Bs,
    int wm, int wn, int lm, int akofs, int bkofs, v8f (&acc)[4][2])
{
    Frag af[4], bfg[2];
    for (int fi = 0; fi < 4; ++fi) {
        const unsigned short* pa = As + (wm + fi * 16 + lm) * KP + akofs;
        af[fi].u[0] = *(const u32x4*)(pa);
        af[fi].u[1] = *(const u32x4*)(pa + 16);
    }
    for (int ni = 0; ni < 2; ++ni) {
        const unsigned short* pb = Bs + (wn + ni * 16 + lm) * KP + bkofs;
        bfg[ni].u[0] = *(const u32x4*)(pb);
        bfg[ni].u[1] = *(const u32x4*)(pb + 16);
    }
    for (int fi = 0; fi < 4; ++fi)
        for (int ni = 0; ni < 2; ++ni)
            acc[fi][ni] = __builtin_amdgcn_wmma_f32_16x16x32_bf16(
                false, af[fi].v, false, bfg[ni].v,
                (short)0, acc[fi][ni], false, false);
}

// ---------------------------------------------------------------------------
// Pre-pass: layout conversion kernels
// ---------------------------------------------------------------------------

// x [b][C][P] f32  ->  xt [b][P][C] bf16   (K-contiguous B^T for GEMM1)
__global__ __launch_bounds__(256) void transpose_x_kernel(
    const float* __restrict__ X, unsigned short* __restrict__ Xt, int C, int P)
{
    __shared__ float tile[32][33];
    int b  = blockIdx.z;
    int p0 = blockIdx.x * 32, c0 = blockIdx.y * 32;
    int tx = threadIdx.x, ty = threadIdx.y;
    for (int r = 0; r < 4; ++r) {
        int c = c0 + ty + r * 8, p = p0 + tx;
        tile[ty + r * 8][tx] = (p < P) ? X[((long long)b * C + c) * P + p] : 0.0f;
    }
    __syncthreads();
    for (int r = 0; r < 4; ++r) {
        int p = p0 + ty + r * 8, c = c0 + tx;
        if (p < P) Xt[((long long)b * P + p) * C + c] = f2bf(tile[tx][ty + r * 8]);
    }
}

// generic fp32 -> bf16 copy
__global__ __launch_bounds__(256) void convert_bf16_kernel(
    const float* __restrict__ in, unsigned short* __restrict__ out, int n)
{
    int i = blockIdx.x * 256 + threadIdx.x;
    if (i < n) out[i] = f2bf(in[i]);
}

// conv_w [O][D][3][3] f32 -> w2p [9][O][D] bf16  (one K-contiguous matrix per tap)
__global__ __launch_bounds__(256) void pack_conv_kernel(
    const float* __restrict__ in, unsigned short* __restrict__ out, int n)
{
    int i = blockIdx.x * 256 + threadIdx.x;
    if (i >= n) return;
    int o = i / (768 * 9);
    int r = i % (768 * 9);
    int d = r / 9, t = r % 9;
    out[((long long)t * 512 + o) * 768 + d] = f2bf(in[i]);
}

// wq/wk/wv/wo [512][512] f32 -> wT[z] [j][k] bf16 (transposed, K-contiguous A)
__global__ __launch_bounds__(256) void transpose_w512_kernel(
    const float* __restrict__ w0, const float* __restrict__ w1,
    const float* __restrict__ w2, const float* __restrict__ w3,
    unsigned short* __restrict__ out)
{
    __shared__ float tile[32][33];
    const float* w = (blockIdx.z == 0) ? w0 : (blockIdx.z == 1) ? w1
                   : (blockIdx.z == 2) ? w2 : w3;
    unsigned short* o = out + (long long)blockIdx.z * 512 * 512;
    int j0 = blockIdx.x * 32, k0 = blockIdx.y * 32;
    int tx = threadIdx.x, ty = threadIdx.y;
    for (int r = 0; r < 4; ++r)
        tile[ty + r * 8][tx] = w[(long long)(k0 + ty + r * 8) * 512 + (j0 + tx)];
    __syncthreads();
    for (int r = 0; r < 4; ++r)
        o[(long long)(j0 + ty + r * 8) * 512 + (k0 + tx)] = f2bf(tile[tx][ty + r * 8]);
}

// ---------------------------------------------------------------------------
// WMMA bf16 GEMM:  C[n][m] = sum_k A[m][k] * Bt[n][k]    (A,Bt K-contiguous)
// 128x128 tile per WG (8 waves, each wave 64M x 32N = 4x2 WMMA tiles),
// K-step 32.  Async path: double-buffered LDS, stage k+1 streams via
// GLOBAL_LOAD_ASYNC_TO_LDS_B128 while WMMAs consume stage k; one barrier +
// one s_wait_asynccnt per K-step.  OOB N rows are simply not staged: their
// columns are never stored by the epilogue.
// EPI: 0 = +bias[m], store bf16   1 = store f32   2 = +bias[m]+res, store f32
// ---------------------------------------------------------------------------
template<int EPI>
__global__ __launch_bounds__(256) void gemm_bf16_kernel(
    const unsigned short* __restrict__ A,
    const unsigned short* __restrict__ Bt,
    float* __restrict__ Cf, unsigned short* __restrict__ Cb,
    const float* __restrict__ bias, const float* __restrict__ res,
    int M, int Ntot, int K, int ldc,
    long long strideB, long long strideC)
{
    __shared__ unsigned short As[2][128 * KP];
    __shared__ unsigned short Bs[2][128 * KP];

    int b  = blockIdx.z;
    const unsigned short* Bb = Bt + (long long)b * strideB;
    int m0 = blockIdx.y * 128, n0 = blockIdx.x * 128;
    int tid = threadIdx.x, wid = tid >> 5, lane = tid & 31;
    int wm = (wid & 1) * 64, wn = (wid >> 1) * 32;

    v8f acc[4][2];
    for (int i = 0; i < 4; ++i)
        for (int j = 0; j < 2; ++j)
            acc[i][j] = v8f{0.f,0.f,0.f,0.f,0.f,0.f,0.f,0.f};

    int lrow = tid >> 1, lhalf = tid & 1;          // cooperative-load assignment
    int akofs = (lane < 16) ? 0 : 8;               // ISA A-frag K layout
    int bkofs = (lane < 16) ? 0 : 8;               // ISA B-frag K layout
    int lm = lane & 15;

    const unsigned short* gA = A + (long long)(m0 + lrow) * K + lhalf * 16;
    int nB = n0 + lrow;
    bool bvalid = (nB < Ntot);
    const unsigned short* gB = Bb + (long long)nB * K + lhalf * 16;
    int ldsoff = lrow * KP + lhalf * 16;
    int KT = K >> 5;

#if HAS_ASYNC_LDS
    { // prologue: stage 0 into buffer 0
        async_cp16(gA,     &As[0][ldsoff]);
        async_cp16(gA + 8, &As[0][ldsoff + 8]);
        if (bvalid) {
            async_cp16(gB,     &Bs[0][ldsoff]);
            async_cp16(gB + 8, &Bs[0][ldsoff + 8]);
        }
    }
    for (int kt = 0; kt < KT; ++kt) {
        int buf = kt & 1;
        async_wait0();          // our async writes into buf have landed in LDS
        __syncthreads();        // everyone's writes landed; prior buf^1 reads retired
        if (kt + 1 < KT) {      // stream stage kt+1 while computing stage kt
            const unsigned short* a1 = gA + (kt + 1) * 32;
            async_cp16(a1,     &As[buf ^ 1][ldsoff]);
            async_cp16(a1 + 8, &As[buf ^ 1][ldsoff + 8]);
            if (bvalid) {
                const unsigned short* b1 = gB + (kt + 1) * 32;
                async_cp16(b1,     &Bs[buf ^ 1][ldsoff]);
                async_cp16(b1 + 8, &Bs[buf ^ 1][ldsoff + 8]);
            }
        }
        frag_mma(As[buf], Bs[buf], wm, wn, lm, akofs, bkofs, acc);
    }
    __syncthreads();
#else
    for (int kt = 0; kt < KT; ++kt) {
        int k0 = kt * 32;
        {
            const u32x4* src = (const u32x4*)(gA + k0);
            u32x4 d0 = src[0], d1 = src[1];
            *(u32x4*)(&As[0][ldsoff])     = d0;
            *(u32x4*)(&As[0][ldsoff + 8]) = d1;
        }
        {
            u32x4 d0 = {0,0,0,0}, d1 = {0,0,0,0};
            if (bvalid) {
                const u32x4* src = (const u32x4*)(gB + k0);
                d0 = src[0]; d1 = src[1];
            }
            *(u32x4*)(&Bs[0][ldsoff])     = d0;
            *(u32x4*)(&Bs[0][ldsoff + 8]) = d1;
        }
        __syncthreads();
        frag_mma(As[0], Bs[0], wm, wn, lm, akofs, bkofs, acc);
        __syncthreads();
    }
#endif

    // epilogue: lane L, reg r -> M = base + r + 8*(L>=16), N = base + (L&15)
    int mL = (lane < 16) ? 0 : 8;
    for (int fi = 0; fi < 4; ++fi) {
        for (int ni = 0; ni < 2; ++ni) {
            int n = n0 + wn + ni * 16 + lm;
            if (n >= Ntot) continue;
            int m = m0 + wm + fi * 16 + mL;
            long long base = (long long)b * strideC + (long long)n * ldc + m;
            if constexpr (EPI == 0) {
                Pack8 pk;
                for (int r = 0; r < 8; ++r)
                    pk.s[r] = f2bf(acc[fi][ni][r] + bias[m + r]);
                *(u32x4*)(Cb + base) = pk.u;
            } else if constexpr (EPI == 1) {
                f32x4 o0, o1;
                for (int r = 0; r < 4; ++r) { o0[r] = acc[fi][ni][r]; o1[r] = acc[fi][ni][r + 4]; }
                *(f32x4*)(Cf + base)     = o0;
                *(f32x4*)(Cf + base + 4) = o1;
            } else {
                f32x4 r0 = *(const f32x4*)(res + base);
                f32x4 r1 = *(const f32x4*)(res + base + 4);
                f32x4 o0, o1;
                for (int r = 0; r < 4; ++r) {
                    o0[r] = acc[fi][ni][r]     + bias[m + r]     + r0[r];
                    o1[r] = acc[fi][ni][r + 4] + bias[m + r + 4] + r1[r];
                }
                *(f32x4*)(Cf + base)     = o0;
                *(f32x4*)(Cf + base + 4) = o1;
            }
        }
    }
}

// per-tap shifted source row for the 3x3 SAME conv
__device__ __forceinline__ bool conv_src(
    int t, int py, int px, int nglob, int bb,
    const unsigned short* F1, const unsigned short** src)
{
    int dy = t / 3 - 1, dx = t % 3 - 1;
    int yy = py + dy, xx = px + dx;
    bool valid = (nglob < 784) && ((unsigned)yy < 28u) && ((unsigned)xx < 28u);
    *src = F1 + ((long long)(bb * 784 + yy * 28 + xx)) * 768;
    return valid;
}

// ---------------------------------------------------------------------------
// 3x3 SAME conv as 9 shifted WMMA GEMMs accumulated in registers.
// Flattened 216-stage pipeline (9 taps x 24 K-steps) so double buffering
// runs across tap boundaries.  Padding rows must be true zeros (they feed
// valid outputs): the consuming thread zeroes its invalid row between the
// async wait and the barrier.
// W2p: [9][512][768] bf16,  F1: [32][784][768] bf16,  F2: [32*784][512] f32
// ---------------------------------------------------------------------------
__global__ __launch_bounds__(256) void conv3x3_wmma_kernel(
    const unsigned short* __restrict__ W2p, const unsigned short* __restrict__ F1,
    const float* __restrict__ bias, float* __restrict__ F2)
{
    __shared__ unsigned short As[2][128 * KP];
    __shared__ unsigned short Bs[2][128 * KP];

    int bb = blockIdx.z;
    int n0 = blockIdx.x * 128, m0 = blockIdx.y * 128;
    int tid = threadIdx.x, wid = tid >> 5, lane = tid & 31;
    int wm = (wid & 1) * 64, wn = (wid >> 1) * 32;

    v8f acc[4][2];
    for (int i = 0; i < 4; ++i)
        for (int j = 0; j < 2; ++j)
            acc[i][j] = v8f{0.f,0.f,0.f,0.f,0.f,0.f,0.f,0.f};

    int lrow = tid >> 1, lhalf = tid & 1;
    int akofs = (lane < 16) ? 0 : 8;
    int bkofs = (lane < 16) ? 0 : 8;
    int lm = lane & 15;

    int nglob = n0 + lrow;
    int py = nglob / 28, px = nglob % 28;
    int ldsoff = lrow * KP + lhalf * 16;
    const int NST = 9 * 24;   // taps x K-steps

#if HAS_ASYNC_LDS
    bool vcons;
    { // prologue: stage 0 (t=0, k0=0)
        const unsigned short* br;
        vcons = conv_src(0, py, px, nglob, bb, F1, &br);
        const unsigned short* ga = W2p + (long long)(m0 + lrow) * 768 + lhalf * 16;
        async_cp16(ga,     &As[0][ldsoff]);
        async_cp16(ga + 8, &As[0][ldsoff + 8]);
        if (vcons) {
            async_cp16(br + lhalf * 16,     &Bs[0][ldsoff]);
            async_cp16(br + lhalf * 16 + 8, &Bs[0][ldsoff + 8]);
        }
    }
    for (int s = 0; s < NST; ++s) {
        int buf = s & 1;
        async_wait0();
        if (!vcons) {   // SAME-padding: force zeros before anyone consumes
            u32x4 z = {0,0,0,0};
            *(u32x4*)(&Bs[buf][ldsoff])     = z;
            *(u32x4*)(&Bs[buf][ldsoff + 8]) = z;
        }
        __syncthreads();
        bool vnext = vcons;
        if (s + 1 < NST) {
            int t1 = (s + 1) / 24, k1 = ((s + 1) % 24) * 32;
            const unsigned short* br;
            vnext = conv_src(t1, py, px, nglob, bb, F1, &br);
            const unsigned short* ga =
                W2p + (long long)t1 * 512 * 768 + (long long)(m0 + lrow) * 768 + k1 + lhalf * 16;
            async_cp16(ga,     &As[buf ^ 1][ldsoff]);
            async_cp16(ga + 8, &As[buf ^ 1][ldsoff + 8]);
            if (vnext) {
                async_cp16(br + k1 + lhalf * 16,     &Bs[buf ^ 1][ldsoff]);
                async_cp16(br + k1 + lhalf * 16 + 8, &Bs[buf ^ 1][ldsoff + 8]);
            }
        }
        frag_mma(As[buf], Bs[buf], wm, wn, lm, akofs, bkofs, acc);
        vcons = vnext;
    }
    __syncthreads();
#else
    for (int s = 0; s < NST; ++s) {
        int t = s / 24, k0 = (s % 24) * 32;
        const unsigned short* br;
        bool valid = conv_src(t, py, px, nglob, bb, F1, &br);
        {
            const u32x4* src = (const u32x4*)(W2p + (long long)t * 512 * 768 +
                                              (long long)(m0 + lrow) * 768 + k0 + lhalf * 16);
            u32x4 d0 = src[0], d1 = src[1];
            *(u32x4*)(&As[0][ldsoff])     = d0;
            *(u32x4*)(&As[0][ldsoff + 8]) = d1;
        }
        {
            u32x4 d0 = {0,0,0,0}, d1 = {0,0,0,0};
            if (valid) {
                const u32x4* src = (const u32x4*)(br + k0 + lhalf * 16);
                d0 = src[0]; d1 = src[1];
            }
            *(u32x4*)(&Bs[0][ldsoff])     = d0;
            *(u32x4*)(&Bs[0][ldsoff + 8]) = d1;
        }
        __syncthreads();
        frag_mma(As[0], Bs[0], wm, wn, lm, akofs, bkofs, acc);
        __syncthreads();
    }
#endif

    int mL = (lane < 16) ? 0 : 8;
    for (int fi = 0; fi < 4; ++fi) {
        for (int ni = 0; ni < 2; ++ni) {
            int n = n0 + wn + ni * 16 + lm;
            if (n >= 784) continue;
            int m = m0 + wm + fi * 16 + mL;
            long long base = ((long long)(bb * 784 + n)) * 512 + m;
            f32x4 o0, o1;
            for (int r = 0; r < 4; ++r) {
                o0[r] = fmaxf(acc[fi][ni][r]     + bias[m + r],     0.0f);
                o1[r] = fmaxf(acc[fi][ni][r + 4] + bias[m + r + 4], 0.0f);
            }
            *(f32x4*)(F2 + base)     = o0;
            *(f32x4*)(F2 + base + 4) = o1;
        }
    }
}

// ---------------------------------------------------------------------------
// Layernorm over 512 channels; one wave per row; writes f32 + bf16 copies
// ---------------------------------------------------------------------------
__global__ __launch_bounds__(256) void layernorm_kernel(
    const float* __restrict__ X, const float* __restrict__ g, const float* __restrict__ bt,
    float* __restrict__ Yf, unsigned short* __restrict__ Yb, int nrows)
{
    int wid = threadIdx.x >> 5, lane = threadIdx.x & 31;
    int row = blockIdx.x * 8 + wid;
    if (row >= nrows) return;
    const float* x = X + (long long)row * 512;
    float v[16], sum = 0.f, sq = 0.f;
    for (int s = 0; s < 4; ++s) {
        f32x4 a = *(const f32x4*)(x + lane * 4 + s * 128);
        for (int e = 0; e < 4; ++e) { float t = a[e]; v[s * 4 + e] = t; sum += t; sq += t * t; }
    }
    sum = wred_sum(sum); sq = wred_sum(sq);
    float mean = sum * (1.f / 512.f);
    float var  = sq * (1.f / 512.f) - mean * mean;
    float rstd = rsqrtf(var + 1e-5f);
    for (int s = 0; s < 4; ++s)
        for (int e = 0; e < 4; ++e) {
            int c = lane * 4 + s * 128 + e;
            float y = (v[s * 4 + e] - mean) * rstd * g[c] + bt[c];
            Yf[(long long)row * 512 + c] = y;
            Yb[(long long)row * 512 + c] = f2bf(y);
        }
}

// ---------------------------------------------------------------------------
// ctx = layernorm(text_features @ proj); one 256-thread block per token row
// ---------------------------------------------------------------------------
__global__ __launch_bounds__(256) void ctx_kernel(
    const float* __restrict__ T, const float* __restrict__ proj,
    const float* __restrict__ g, const float* __restrict__ bt,
    unsigned short* __restrict__ Cb)
{
    __shared__ float tr[768];
    __shared__ float outv[512];
    __shared__ float red[18];
    long long row = blockIdx.x;
    int tid = threadIdx.x, wid = tid >> 5, lane = tid & 31;
    for (int i = tid; i < 768; i += 256) tr[i] = T[row * 768 + i];
    __syncthreads();
    float s = 0.f, sq = 0.f;
    for (int r = 0; r < 2; ++r) {
        int o = tid + r * 256;
        float acc = 0.f;
        for (int k = 0; k < 768; ++k) acc = fmaf(tr[k], proj[(long long)k * 512 + o], acc);
        outv[o] = acc; s += acc; sq += acc * acc;
    }
    s = wred_sum(s); sq = wred_sum(sq);
    if (lane == 0) { red[wid] = s; red[8 + wid] = sq; }
    __syncthreads();
    if (tid == 0) {
        float ts = 0.f, tq = 0.f;
        for (int i = 0; i < 8; ++i) { ts += red[i]; tq += red[8 + i]; }
        red[16] = ts; red[17] = tq;
    }
    __syncthreads();
    float mean = red[16] * (1.f / 512.f);
    float var  = red[17] * (1.f / 512.f) - mean * mean;
    float rstd = rsqrtf(var + 1e-5f);
    for (int r = 0; r < 2; ++r) {
        int o = tid + r * 256;
        float y = (outv[o] - mean) * rstd * g[o] + bt[o];
        Cb[row * 512 + o] = f2bf(y);
    }
}

// ---------------------------------------------------------------------------
// Cross-attention: block = (i, b), wave = head. 77 keys -> one-pass softmax.
// ---------------------------------------------------------------------------
__global__ __launch_bounds__(256) void attention_kernel(
    const float* __restrict__ Q, const float* __restrict__ Kc,
    const float* __restrict__ Vc, unsigned short* __restrict__ Obf)
{
    __shared__ float probs[8][80];
    int i = blockIdx.x, b = blockIdx.y;
    int h = threadIdx.x >> 5, lane = threadIdx.x & 31;
    const float* q = Q + ((long long)(b * 784 + i)) * 512 + h * 64;

    float s[3];
    for (int jj = 0; jj < 3; ++jj) {
        int j = lane + jj * 32;
        float acc;
        if (j < 77) {
            const float* kr = Kc + ((long long)(b * 77 + j)) * 512 + h * 64;
            acc = 0.f;
            for (int d = 0; d < 64; d += 4) {
                f32x4 qa = *(const f32x4*)(q + d);
                f32x4 ka = *(const f32x4*)(kr + d);
                for (int e = 0; e < 4; ++e) acc = fmaf(qa[e], ka[e], acc);
            }
            acc *= 0.125f;                      // dim_head^-0.5
        } else acc = -INFINITY;
        s[jj] = acc;
    }
    float m = fmaxf(fmaxf(s[0], s[1]), s[2]);
    m = wred_max(m);
    float p[3], sum = 0.f;
    for (int jj = 0; jj < 3; ++jj) { p[jj] = __expf(s[jj] - m); if (s[jj] == -INFINITY) p[jj] = 0.f; sum += p[jj]; }
    sum = wred_sum(sum);
    float inv = 1.f / sum;
    probs[h][lane]      = p[0] * inv;
    probs[h][lane + 32] = p[1] * inv;
    if (lane < 16) probs[h][lane + 64] = p[2] * inv;
    __syncthreads();

    for (int dd = 0; dd < 64; dd += 32) {
        int d = lane + dd;
        float acc = 0.f;
        for (int j = 0; j < 77; ++j)
            acc = fmaf(probs[h][j], Vc[((long long)(b * 77 + j)) * 512 + h * 64 + d], acc);
        Obf[((long long)(b * 784 + i)) * 512 + h * 64 + d] = f2bf(acc);
    }
}

// ---------------------------------------------------------------------------
// Pool + gating + 4 tiny experts + top-3 combine; one block per batch
// ---------------------------------------------------------------------------
__global__ __launch_bounds__(256) void final_kernel(
    const float* __restrict__ Ff, const float* __restrict__ gate_w,
    const float* __restrict__ gate_b, const float* __restrict__ ew1,
    const float* __restrict__ eb1, const float* __restrict__ ew2,
    const float* __restrict__ eb2, float* __restrict__ out)
{
    __shared__ float pooled[512];
    __shared__ float red[256];
    __shared__ float gates[4];
    __shared__ float eout[4];
    int b = blockIdx.x, tid = threadIdx.x;

    for (int r = 0; r < 2; ++r) {
        int c = tid + r * 256;
        float s = 0.f;
        for (int i = 0; i < 784; ++i) s += Ff[((long long)(b * 784 + i)) * 512 + c];
        pooled[c] = s * (1.f / 784.f);
    }
    __syncthreads();
    if (tid < 4) {
        float a = gate_b[tid];
        for (int c = 0; c < 512; ++c) a = fmaf(pooled[c], gate_w[c * 4 + tid], a);
        gates[tid] = a;
    }
    __syncthreads();
    if (tid == 0) {
        float m = fmaxf(fmaxf(gates[0], gates[1]), fmaxf(gates[2], gates[3]));
        float s = 0.f;
        for (int e = 0; e < 4; ++e) { gates[e] = expf(gates[e] - m); s += gates[e]; }
        for (int e = 0; e < 4; ++e) gates[e] /= s;
    }
    __syncthreads();

    for (int e = 0; e < 4; ++e) {
        float part = 0.f;
        for (int r = 0; r < 2; ++r) {
            int h = tid + r * 256;
            float a = eb1[e * 512 + h];
            const float* wcol = ew1 + ((long long)e * 512) * 512 + h;
            for (int d = 0; d < 512; ++d) a = fmaf(pooled[d], wcol[(long long)d * 512], a);
            float gel = 0.5f * a * (1.f + erff(a * 0.70710678118654752f));  // exact gelu
            part = fmaf(gel, ew2[e * 512 + h], part);
        }
        red[tid] = part;
        __syncthreads();
        for (int sft = 128; sft > 0; sft >>= 1) {
            if (tid < sft) red[tid] += red[tid + sft];
            __syncthreads();
        }
        if (tid == 0) eout[e] = red[0] + eb2[e];
        __syncthreads();
    }

    if (tid == 0) {
        // top-3 of 4 == exclude the smallest gate
        int mn = 0;
        for (int e = 1; e < 4; ++e) if (gates[e] < gates[mn]) mn = e;
        float r = 0.f;
        for (int e = 0; e < 4; ++e) if (e != mn) r = fmaf(gates[e], eout[e], r);
        out[b] = r;
    }
}

// ---------------------------------------------------------------------------
// Host-side launch
// ---------------------------------------------------------------------------
extern "C" void kernel_launch(void* const* d_in, const int* in_sizes, int n_in,
                              void* d_out, int out_size, void* d_ws, size_t ws_size,
                              hipStream_t stream)
{
    (void)in_sizes; (void)n_in; (void)out_size; (void)ws_size;
    const float* x      = (const float*)d_in[0];
    const float* textf  = (const float*)d_in[1];
    const float* dc_w   = (const float*)d_in[2];
    const float* dc_b   = (const float*)d_in[3];
    const float* conv_w = (const float*)d_in[4];
    const float* conv_b = (const float*)d_in[5];
    const float* n1_g   = (const float*)d_in[6];
    const float* n1_b   = (const float*)d_in[7];
    const float* n2_g   = (const float*)d_in[8];
    const float* n2_b   = (const float*)d_in[9];
    const float* proj   = (const float*)d_in[10];
    const float* wq     = (const float*)d_in[11];
    const float* wk     = (const float*)d_in[12];
    const float* wv     = (const float*)d_in[13];
    const float* wo     = (const float*)d_in[14];
    const float* wo_b   = (const float*)d_in[15];
    const float* gate_w = (const float*)d_in[16];
    const float* gate_b = (const float*)d_in[17];
    const float* ew1    = (const float*)d_in[18];
    const float* eb1    = (const float*)d_in[19];
    const float* ew2    = (const float*)d_in[20];
    const float* eb2    = (const float*)d_in[21];
    float* out = (float*)d_out;

    constexpr long long B = 32, P = 784, CIN = 3072, D = 768, O = 512, T = 77;
    constexpr long long ROWS = B * P;        // 25088
    constexpr long long TROWS = B * T;       // 2464

    char* wsb = (char*)d_ws;
    size_t off = 0;
    auto alloc = [&](size_t bytes) -> char* {
        char* p = wsb + off;
        off += (bytes + 255) & ~(size_t)255;
        return p;
    };
    unsigned short* xt    = (unsigned short*)alloc(B * P * CIN * 2);   // x^T bf16
    unsigned short* dcw   = (unsigned short*)alloc(D * CIN * 2);       // dc_w bf16
    unsigned short* w2p   = (unsigned short*)alloc(9 * O * D * 2);     // conv taps
    unsigned short* wT    = (unsigned short*)alloc(4 * O * O * 2);     // wq/wk/wv/wo^T
    unsigned short* f1    = (unsigned short*)alloc(B * P * D * 2);     // conv1 out bf16 [b][p][d]
    float*          f2    = (float*)alloc(ROWS * O * 4);               // conv2 out f32
    float*          fn_f  = (float*)alloc(ROWS * O * 4);               // layernorm f32
    unsigned short* fn_bf = (unsigned short*)alloc(ROWS * O * 2);
    unsigned short* ctx_b = (unsigned short*)alloc(TROWS * O * 2);
    float*          q_f   = (float*)alloc(ROWS * O * 4);
    float*          k_f   = (float*)alloc(TROWS * O * 4);
    float*          v_f   = (float*)alloc(TROWS * O * 4);
    unsigned short* ao_bf = (unsigned short*)alloc(ROWS * O * 2);
    float*          ff_f  = (float*)alloc(ROWS * O * 4);

    // --- pre-pass conversions ---
    transpose_x_kernel<<<dim3(25, 96, 32), dim3(32, 8), 0, stream>>>(x, xt, (int)CIN, (int)P);
    {
        int n = (int)(D * CIN);
        convert_bf16_kernel<<<(n + 255) / 256, 256, 0, stream>>>(dc_w, dcw, n);
    }
    {
        int n = (int)(O * D * 9);
        pack_conv_kernel<<<(n + 255) / 256, 256, 0, stream>>>(conv_w, w2p, n);
    }
    transpose_w512_kernel<<<dim3(16, 16, 4), dim3(32, 8), 0, stream>>>(wq, wk, wv, wo, wT);

    // --- 1x1 conv: f1[b][p][d] = x[b][:,p] . dc_w[d][:] + dc_b  (bf16 out) ---
    gemm_bf16_kernel<0><<<dim3(7, 6, 32), 256, 0, stream>>>(
        dcw, xt, nullptr, f1, dc_b, nullptr,
        (int)D, (int)P, (int)CIN, (int)D, P * CIN, P * D);

    // --- 3x3 conv + relu -> f2 ---
    conv3x3_wmma_kernel<<<dim3(7, 4, 32), 256, 0, stream>>>(w2p, f1, conv_b, f2);

    // --- layernorm(f) -> fn_f (residual) + fn_bf (GEMM operand) ---
    layernorm_kernel<<<(int)(ROWS / 8), 256, 0, stream>>>(f2, n1_g, n1_b, fn_f, fn_bf, (int)ROWS);

    // --- ctx = layernorm(text @ proj) -> bf16 ---
    ctx_kernel<<<(int)TROWS, 256, 0, stream>>>(textf, proj, n2_g, n2_b, ctx_b);

    // --- q / k / v projections (WMMA) ---
    gemm_bf16_kernel<1><<<dim3(196, 4, 1), 256, 0, stream>>>(
        wT + 0 * O * O, fn_bf, q_f, nullptr, nullptr, nullptr,
        (int)O, (int)ROWS, (int)O, (int)O, 0, 0);
    gemm_bf16_kernel<1><<<dim3(20, 4, 1), 256, 0, stream>>>(
        wT + 1 * O * O, ctx_b, k_f, nullptr, nullptr, nullptr,
        (int)O, (int)TROWS, (int)O, (int)O, 0, 0);
    gemm_bf16_kernel<1><<<dim3(20, 4, 1), 256, 0, stream>>>(
        wT + 2 * O * O, ctx_b, v_f, nullptr, nullptr, nullptr,
        (int)O, (int)TROWS, (int)O, (int)O, 0, 0);

    // --- attention ---
    attention_kernel<<<dim3(784, 32), 256, 0, stream>>>(q_f, k_f, v_f, ao_bf);

    // --- out projection + bias + residual (WMMA) ---
    gemm_bf16_kernel<2><<<dim3(196, 4, 1), 256, 0, stream>>>(
        wT + 3 * O * O, ao_bf, ff_f, nullptr, wo_b, fn_f,
        (int)O, (int)ROWS, (int)O, (int)O, 0, 0);

    // --- pool + gate + experts ---
    final_kernel<<<32, 256, 0, stream>>>(ff_f, gate_w, gate_b, ew1, eb1, ew2, eb2, out);
}
